// StreamAttention_8701603741985
// MI455X (gfx1250) — matrix operand order repaired and checked by reading, not compiled, not run on hardware
//
#include <hip/hip_runtime.h>
#include <hip/hip_bf16.h>
#include <stdint.h>

typedef __attribute__((ext_vector_type(16))) _Float16 v16h;
typedef __attribute__((ext_vector_type(8)))  _Float16 v8h;
typedef __attribute__((ext_vector_type(8)))  float    v8f;
typedef __attribute__((ext_vector_type(4)))  uint32_t u32x4;
typedef __attribute__((ext_vector_type(8)))  uint32_t u32x8;

#define HIDDEN 2048
#define NHEADS 16
#define HDIM   128
#define BATCH  4
#define SEQ    2048
#define MROWS  (BATCH * SEQ)  // 8192
#define LN_EPS 1e-5f

// ---------------------------------------------------------------------------
// WMMA fragment helpers (CDNA5 16x16x32 f16 layouts, CDNA5 ISA)
// A 16x32 (MxK):  lane L: row = L&15, kbase = (L<16)?0:8,
//                 elems 0..7 -> K = kbase+e, elems 8..15 -> K = kbase+16+(e-8)
// B 32x16 (KxN):  lane L: col = L&15, same K mapping as A (kbase = (L<16)?0:8)
// C/D 16x16 f32:  lane L: col = L&15, VGPR v: row = v + ((L<16)?0:8)
// ---------------------------------------------------------------------------
__device__ __forceinline__ v16h load_frag_a(const _Float16* base, int lane, int ld) {
  const int row = lane & 15;
  const int kb  = (lane < 16) ? 0 : 8;
  const _Float16* p = base + row * ld + kb;
  v8h lo = *(const v8h*)(p);
  v8h hi = *(const v8h*)(p + 16);
  v16h r;
#pragma unroll
  for (int i = 0; i < 8; ++i) { r[i] = lo[i]; r[i + 8] = hi[i]; }
  return r;
}

__device__ __forceinline__ v16h load_frag_b(const _Float16* base, int lane, int ld) {
  const int row = lane & 15;  // N index
  const int kb  = (lane < 16) ? 0 : 8;
  const _Float16* p = base + row * ld + kb;
  v8h lo = *(const v8h*)(p);
  v8h hi = *(const v8h*)(p + 16);
  v16h r;
#pragma unroll
  for (int i = 0; i < 8; ++i) { r[i] = lo[i]; r[i + 8] = hi[i]; }
  return r;
}

__device__ __forceinline__ v8f wmma_f16(v16h a, v16h b, v8f c) {
  return __builtin_amdgcn_wmma_f32_16x16x32_f16(false, a, false, b, (short)0, c,
                                                false, false);
}

// ---------------------------------------------------------------------------
// Tensor Data Mover: DMA a [rows=128 x cols=32] f16 tile (row stride 2048
// elements in global) into LDS, with HW padding of 4 DWORDs (8 halves) after
// every 16 DWORDs (32 halves) -> LDS row pitch of 40 halves, matching the
// fragment loaders. D# layout per CDNA5 ISA.
// One instruction per wave; tracked by TENSORcnt.
// ---------------------------------------------------------------------------
__device__ __forceinline__ void tdm_load_tile_128x32(const _Float16* gsrc,
                                                     uint32_t lds_byte_off) {
  const uint64_t ga = (uint64_t)(size_t)gsrc;
  u32x4 g0;
  g0[0] = 1u;                                            // count=1, user desc
  g0[1] = lds_byte_off;                                  // lds_addr [63:32]
  g0[2] = (uint32_t)ga;                                  // global_addr lo
  g0[3] = (uint32_t)((ga >> 32) & 0x01FFFFFFu) | (2u << 30);  // addr hi | type=2
  u32x8 g1;
  // data_size=1 (2B) | pad_enable | pad_interval=3 (16 DW) | pad_amount=3 (4 DW)
  g1[0] = (1u << 16) | (1u << 20) | (3u << 22) | (3u << 25);
  g1[1] = (32u << 16);    // tensor_dim0 = 32 (low 16 bits at [63:48])
  g1[2] = (128u << 16);   // tensor_dim0 hi=0, tensor_dim1 = 128
  g1[3] = (32u << 16);    // tensor_dim1 hi=0, tile_dim0 = 32
  g1[4] = 128u;           // tile_dim1 = 128, tile_dim2 = 0
  g1[5] = 2048u;          // tensor_dim0_stride = 2048 elements
  g1[6] = 0u;             // stride0 hi, tensor_dim1_stride lo (unused, 2D)
  g1[7] = 0u;
  asm volatile("tensor_load_to_lds %0, %1" ::"s"(g0), "s"(g1) : "memory");
}

__device__ __forceinline__ uint32_t lds_offset_of(const void* p) {
  // generic LDS address: addr[31:0] is the LDS byte offset (ISA §10.2)
  return (uint32_t)(size_t)p;
}

// ---------------------------------------------------------------------------
// fp32 -> f16 cast (grid-stride)
// ---------------------------------------------------------------------------
__global__ void cast_f32_to_f16_kernel(const float* __restrict__ in,
                                       _Float16* __restrict__ out, int n) {
  int i = blockIdx.x * blockDim.x + threadIdx.x;
  const int stride = gridDim.x * blockDim.x;
  for (; i < n; i += stride) out[i] = (_Float16)in[i];
}

// ---------------------------------------------------------------------------
// C[M,N] = A[M,K] @ W[N,K]^T + bias, M=8192, N=K=2048.
// Block: 256 thr (8 waves), 128x128 tile, K-step 32. Tiles staged by the
// Tensor Data Mover into double-buffered LDS (wave 0 issues 2 descriptors per
// step; TDM of tile t+1 overlaps WMMA on tile t; fenced by s_wait_tensorcnt +
// workgroup barrier). Waves 4(M) x 2(N); each wave computes 2x4 16x16 tiles.
// SPLIT_HEADS: f16 out scattered to [B][H][S][D] (with scale folded in);
// else fp32 row-major [M][N] (output projection into d_out).
// ---------------------------------------------------------------------------
template <bool SPLIT_HEADS>
__global__ void __launch_bounds__(256)
gemm_wmma_kernel(const _Float16* __restrict__ A, const _Float16* __restrict__ W,
                 const float* __restrict__ bias, void* __restrict__ Cout,
                 float scale) {
  __shared__ alignas(16) _Float16 As[2][128 * 40];
  __shared__ alignas(16) _Float16 Ws[2][128 * 40];

  const int tid  = threadIdx.x;
  const int lane = tid & 31;
  const int wid  = tid >> 5;
  const int wm   = wid & 3;   // M offset wm*32
  const int wn   = wid >> 2;  // N offset wn*64
  const int mblock = blockIdx.y * 128;
  const int nblock = blockIdx.x * 128;

  const _Float16* Abase = A + (size_t)mblock * HIDDEN;
  const _Float16* Wbase = W + (size_t)nblock * HIDDEN;
  const uint32_t asOff0 = lds_offset_of(&As[0][0]);
  const uint32_t asOff1 = lds_offset_of(&As[1][0]);
  const uint32_t wsOff0 = lds_offset_of(&Ws[0][0]);
  const uint32_t wsOff1 = lds_offset_of(&Ws[1][0]);

  v8f acc[2][4];
#pragma unroll
  for (int i = 0; i < 2; ++i)
#pragma unroll
    for (int j = 0; j < 4; ++j)
#pragma unroll
      for (int e = 0; e < 8; ++e) acc[i][j][e] = 0.f;

  // Preload tile 0
  if (wid == 0) {
    tdm_load_tile_128x32(Abase, asOff0);
    tdm_load_tile_128x32(Wbase, wsOff0);
    __builtin_amdgcn_s_wait_tensorcnt(0);
  }
  __syncthreads();

  const int NT = HIDDEN / 32;  // 64 K-steps
  for (int t = 0; t < NT; ++t) {
    const int cur = t & 1;
    // Kick off DMA for next tile while computing on the current one.
    if (t + 1 < NT && wid == 0) {
      const int koff = (t + 1) * 32;
      tdm_load_tile_128x32(Abase + koff, cur ? asOff0 : asOff1);
      tdm_load_tile_128x32(Wbase + koff, cur ? wsOff0 : wsOff1);
    }

    const _Float16* curA = As[cur];
    const _Float16* curW = Ws[cur];
    v16h af[2], bf[4];
#pragma unroll
    for (int mt = 0; mt < 2; ++mt)
      af[mt] = load_frag_a(curA + (wm * 32 + mt * 16) * 40, lane, 40);
#pragma unroll
    for (int nt = 0; nt < 4; ++nt)
      bf[nt] = load_frag_b(curW + (wn * 64 + nt * 16) * 40, lane, 40);
#pragma unroll
    for (int mt = 0; mt < 2; ++mt)
#pragma unroll
      for (int nt = 0; nt < 4; ++nt)
        acc[mt][nt] = wmma_f16(af[mt], bf[nt], acc[mt][nt]);

    __syncthreads();                 // everyone done reading buffers
    if (wid == 0) __builtin_amdgcn_s_wait_tensorcnt(0);  // next tile landed
    __syncthreads();                 // publish next buffer to all waves
  }

  const int mbase = (lane < 16) ? 0 : 8;
#pragma unroll
  for (int mt = 0; mt < 2; ++mt) {
#pragma unroll
    for (int nt = 0; nt < 4; ++nt) {
      const int n = nblock + wn * 64 + nt * 16 + (lane & 15);
      const float bv = bias[n];
#pragma unroll
      for (int v = 0; v < 8; ++v) {
        const int m = mblock + wm * 32 + mt * 16 + mbase + v;
        const float val = (acc[mt][nt][v] + bv) * scale;
        if (SPLIT_HEADS) {
          const int b = m >> 11, s = m & 2047;   // m = b*SEQ + s
          const int h = n >> 7,  d = n & 127;    // n = h*HDIM + d
          ((_Float16*)Cout)[(((size_t)b * NHEADS + h) * SEQ + s) * HDIM + d] =
              (_Float16)val;
        } else {
          ((float*)Cout)[(size_t)m * HIDDEN + n] = val;
        }
      }
    }
  }
}

// ---------------------------------------------------------------------------
// Causal flash attention. Grid: (SEQ/64, NHEADS, BATCH), block 128 (4 waves).
// Each wave owns 16 query rows; KV processed in tiles of 32 keys shared via
// LDS (K as [j][d] = B-layout for QK^T; V transposed to [d][j] = B-layout for
// PV). Softmax stats live replicated per 16-lane half; P transits a per-wave
// LDS scratch to become an A-fragment.
// ---------------------------------------------------------------------------
__global__ void __launch_bounds__(128)
flash_attn_kernel(const _Float16* __restrict__ Q, const _Float16* __restrict__ K,
                  const _Float16* __restrict__ V, _Float16* __restrict__ Out) {
  __shared__ alignas(16) _Float16 Ks[32 * 136];
  __shared__ alignas(16) _Float16 Vs[128 * 40];
  __shared__ alignas(16) _Float16 Pl[4 * 16 * 40];

  const int tid    = threadIdx.x;
  const int lane   = tid & 31;
  const int wid    = tid >> 5;
  const int qblock = blockIdx.x * 64;
  const int h      = blockIdx.y;
  const int b      = blockIdx.z;
  const size_t bh  = (size_t)b * NHEADS + h;

  const int qrow0 = qblock + wid * 16;
  const _Float16* qbase = Q + (bh * SEQ + qrow0) * HDIM;

  v16h qf[4];
#pragma unroll
  for (int dk = 0; dk < 4; ++dk)
    qf[dk] = load_frag_a(qbase + dk * 32, lane, HDIM);

  float mrow[8], lrow[8];
  v8f o[8];
#pragma unroll
  for (int v = 0; v < 8; ++v) { mrow[v] = -3.0e38f; lrow[v] = 0.f; }
#pragma unroll
  for (int dt = 0; dt < 8; ++dt)
#pragma unroll
    for (int e = 0; e < 8; ++e) o[dt][e] = 0.f;

  const int jload = tid >> 2;        // 0..31  (key row staged)
  const int dload = (tid & 3) * 32;  // d chunk staged
  const int jj = lane & 15;
  const int mb = (lane < 16) ? 0 : 8;
  _Float16* Pw = Pl + wid * (16 * 40);

  const int nkv = (qblock >> 5) + 2;  // causal: keys up to qblock+63
  for (int t = 0; t < nkv; ++t) {
    const int kv0 = t * 32;
    __syncthreads();
    {
      const _Float16* ksrc = K + (bh * SEQ + kv0 + jload) * HDIM + dload;
#pragma unroll
      for (int c = 0; c < 4; ++c)
        *(v8h*)(Ks + jload * 136 + dload + c * 8) = *(const v8h*)(ksrc + c * 8);
      const _Float16* vsrc = V + (bh * SEQ + kv0 + jload) * HDIM + dload;
#pragma unroll
      for (int c = 0; c < 4; ++c) {
        v8h vv = *(const v8h*)(vsrc + c * 8);
#pragma unroll
        for (int e = 0; e < 8; ++e)
          Vs[(dload + c * 8 + e) * 40 + jload] = vv[e];  // transpose to [d][j]
      }
    }
    __syncthreads();

    v8f s0, s1;
#pragma unroll
    for (int e = 0; e < 8; ++e) { s0[e] = 0.f; s1[e] = 0.f; }
#pragma unroll
    for (int dk = 0; dk < 4; ++dk) {
      v16h kf0 = load_frag_b(Ks + dk * 32, lane, 136);
      v16h kf1 = load_frag_b(Ks + 16 * 136 + dk * 32, lane, 136);
      s0 = wmma_f16(qf[dk], kf0, s0);
      s1 = wmma_f16(qf[dk], kf1, s1);
    }

    float alpha[8];
#pragma unroll
    for (int v = 0; v < 8; ++v) {
      const int rowg = qrow0 + mb + v;
      float a0 = (kv0 + jj      <= rowg) ? s0[v] : -3.0e38f;
      float a1 = (kv0 + jj + 16 <= rowg) ? s1[v] : -3.0e38f;
      float mt = fmaxf(a0, a1);
#pragma unroll
      for (int off = 1; off < 16; off <<= 1)
        mt = fmaxf(mt, __shfl_xor(mt, off, 32));
      const float mnew = fmaxf(mrow[v], mt);
      const float al = __expf(mrow[v] - mnew);
      const float p0 = __expf(a0 - mnew);
      const float p1 = __expf(a1 - mnew);
      float ls = p0 + p1;
#pragma unroll
      for (int off = 1; off < 16; off <<= 1) ls += __shfl_xor(ls, off, 32);
      lrow[v] = lrow[v] * al + ls;
      mrow[v] = mnew;
      alpha[v] = al;
      Pw[(mb + v) * 40 + jj]      = (_Float16)p0;
      Pw[(mb + v) * 40 + jj + 16] = (_Float16)p1;
    }
#pragma unroll
    for (int dt = 0; dt < 8; ++dt)
#pragma unroll
      for (int v = 0; v < 8; ++v) o[dt][v] *= alpha[v];

    v16h pa = load_frag_a(Pw, lane, 40);  // P 16x32 as A-fragment
#pragma unroll
    for (int dt = 0; dt < 8; ++dt) {
      v16h vf = load_frag_b(Vs + dt * 16 * 40, lane, 40);
      o[dt] = wmma_f16(pa, vf, o[dt]);
    }
  }

#pragma unroll
  for (int v = 0; v < 8; ++v) lrow[v] = 1.0f / lrow[v];
#pragma unroll
  for (int dt = 0; dt < 8; ++dt) {
    const int d = dt * 16 + jj;
#pragma unroll
    for (int v = 0; v < 8; ++v) {
      const int rowg = qrow0 + mb + v;
      Out[(size_t)(b * SEQ + rowg) * HIDDEN + h * HDIM + d] =
          (_Float16)(o[dt][v] * lrow[v]);
    }
  }
}

// ---------------------------------------------------------------------------
// In-place LayerNorm over HIDDEN=2048, one block (256 thr) per row.
// ---------------------------------------------------------------------------
__global__ void __launch_bounds__(256)
layernorm_kernel(float* __restrict__ out, const float* __restrict__ gamma,
                 const float* __restrict__ beta) {
  __shared__ float red[8];
  const int tid = threadIdx.x;
  float* p = out + (size_t)blockIdx.x * HIDDEN;

  float x[8];
  float s = 0.f;
#pragma unroll
  for (int i = 0; i < 8; ++i) { x[i] = p[tid + i * 256]; s += x[i]; }
#pragma unroll
  for (int off = 1; off < 32; off <<= 1) s += __shfl_xor(s, off, 32);
  if ((tid & 31) == 0) red[tid >> 5] = s;
  __syncthreads();
  float mu = 0.f;
#pragma unroll
  for (int i = 0; i < 8; ++i) mu += red[i];
  mu *= (1.0f / HIDDEN);
  __syncthreads();

  float q = 0.f;
#pragma unroll
  for (int i = 0; i < 8; ++i) { const float d = x[i] - mu; q += d * d; }
#pragma unroll
  for (int off = 1; off < 32; off <<= 1) q += __shfl_xor(q, off, 32);
  if ((tid & 31) == 0) red[tid >> 5] = q;
  __syncthreads();
  float var = 0.f;
#pragma unroll
  for (int i = 0; i < 8; ++i) var += red[i];
  var *= (1.0f / HIDDEN);
  const float rs = rsqrtf(var + LN_EPS);

#pragma unroll
  for (int i = 0; i < 8; ++i) {
    const int idx = tid + i * 256;
    p[idx] = (x[i] - mu) * rs * gamma[idx] + beta[idx];
  }
}

// ---------------------------------------------------------------------------
extern "C" void kernel_launch(void* const* d_in, const int* in_sizes, int n_in,
                              void* d_out, int out_size, void* d_ws,
                              size_t ws_size, hipStream_t stream) {
  const float* hs    = (const float*)d_in[0];
  const float* Wq    = (const float*)d_in[1];
  const float* bq    = (const float*)d_in[2];
  const float* Wk    = (const float*)d_in[3];
  const float* bk    = (const float*)d_in[4];
  const float* Wv    = (const float*)d_in[5];
  const float* bv    = (const float*)d_in[6];
  const float* Wo    = (const float*)d_in[7];
  const float* bo    = (const float*)d_in[8];
  const float* gamma = (const float*)d_in[9];
  const float* beta  = (const float*)d_in[10];

  char* ws = (char*)d_ws;
  size_t off = 0;
  auto carve = [&](size_t bytes) {
    char* p = ws + off;
    off += (bytes + 255) & ~(size_t)255;
    return p;
  };
  const size_t actB = (size_t)MROWS * HIDDEN * sizeof(_Float16);   // 32 MiB
  const size_t wB   = (size_t)HIDDEN * HIDDEN * sizeof(_Float16);  // 8 MiB
  _Float16* hs16   = (_Float16*)carve(actB);
  _Float16* wq16   = (_Float16*)carve(wB);
  _Float16* wk16   = (_Float16*)carve(wB);
  _Float16* wv16   = (_Float16*)carve(wB);
  _Float16* wo16   = (_Float16*)carve(wB);
  _Float16* q16    = (_Float16*)carve(actB);
  _Float16* k16    = (_Float16*)carve(actB);
  _Float16* v16    = (_Float16*)carve(actB);
  _Float16* attn16 = (_Float16*)carve(actB);

  // 1) fp32 -> f16 casts
  cast_f32_to_f16_kernel<<<4096, 256, 0, stream>>>(hs, hs16, MROWS * HIDDEN);
  cast_f32_to_f16_kernel<<<2048, 256, 0, stream>>>(Wq, wq16, HIDDEN * HIDDEN);
  cast_f32_to_f16_kernel<<<2048, 256, 0, stream>>>(Wk, wk16, HIDDEN * HIDDEN);
  cast_f32_to_f16_kernel<<<2048, 256, 0, stream>>>(Wv, wv16, HIDDEN * HIDDEN);
  cast_f32_to_f16_kernel<<<2048, 256, 0, stream>>>(Wo, wo16, HIDDEN * HIDDEN);

  // 2) QKV projections (1/sqrt(HEAD_DIM) folded into Q)
  const dim3 gg(HIDDEN / 128, MROWS / 128);
  const float qscale = 0.08838834764831845f;  // 1/sqrt(128)
  gemm_wmma_kernel<true><<<gg, 256, 0, stream>>>(hs16, wq16, bq, q16, qscale);
  gemm_wmma_kernel<true><<<gg, 256, 0, stream>>>(hs16, wk16, bk, k16, 1.0f);
  gemm_wmma_kernel<true><<<gg, 256, 0, stream>>>(hs16, wv16, bv, v16, 1.0f);

  // 3) causal flash attention -> attn16 [B*S][HIDDEN]
  flash_attn_kernel<<<dim3(SEQ / 64, NHEADS, BATCH), 128, 0, stream>>>(
      q16, k16, v16, attn16);

  // 4) output projection (fp32 into d_out), then in-place LayerNorm
  gemm_wmma_kernel<false><<<gg, 256, 0, stream>>>(attn16, wo16, bo, d_out, 1.0f);
  layernorm_kernel<<<MROWS, 256, 0, stream>>>((float*)d_out, gamma, beta);
}
